// RNN_28887950033086
// MI455X (gfx1250) — matrix operand order repaired and checked by reading, not compiled, not run on hardware
//
#include <hip/hip_runtime.h>

#define S_LEN 2048
#define BATCH 32
#define NIN   64
#define NHID  512
#define NOUT  64
#define NWG   16
#define NTH   256
#define XLD   (S_LEN * NIN)   // row stride of x per batch row
#define HBUF_ELEMS (BATCH * NHID)

typedef _Float16 f16;
typedef __attribute__((ext_vector_type(16))) _Float16 v16h;
typedef __attribute__((ext_vector_type(8)))  float    v8f;

#define WMMA16(a, b, c) \
  __builtin_amdgcn_wmma_f32_16x16x32_f16(false, (a), false, (b), (short)0, (c), false, false)

union V16U { v16h v; unsigned u[8]; _Float16 e[16]; };

// ---- A-operand pack: 16x32 f16 tile, per ISA 16-bit A layout ----
// lanes 0-15: v0..3 -> K 0..7, v4..7 -> K 16..23 ; lanes 16-31: K 8..15 / 24..31
__device__ __forceinline__ v16h loadA_f16(const f16* __restrict__ p, int ld, int kbase, int lane) {
  const int m = lane & 15, hf = lane >> 4;
  V16U r;
  const f16* row = p + (size_t)m * ld + kbase;
#pragma unroll
  for (int q = 0; q < 8; ++q) {
    const int k0 = (q < 4) ? (hf * 8 + 2 * q) : (16 + hf * 8 + 2 * (q - 4));
    r.u[q] = *(const unsigned*)(row + k0);
  }
  return r.v;
}

__device__ __forceinline__ v16h loadA_f32(const float* __restrict__ p, int ld, int kbase, int lane) {
  const int m = lane & 15, hf = lane >> 4;
  V16U r;
  const float* row = p + (size_t)m * ld + kbase;
#pragma unroll
  for (int q = 0; q < 8; ++q) {
    const int k0 = (q < 4) ? (hf * 8 + 2 * q) : (16 + hf * 8 + 2 * (q - 4));
    r.e[2 * q]     = (_Float16)row[k0];
    r.e[2 * q + 1] = (_Float16)row[k0 + 1];
  }
  return r.v;
}

// ---- B-operand pack: 32x16 f16, B[k][n] = W[nbase+n][kbase+k] (W row-major) ----
__device__ __forceinline__ v16h loadB_f16(const f16* __restrict__ w, int ldw, int nbase,
                                          int kbase, int lane) {
  const int n = lane & 15, hf = lane >> 4;
  V16U r;
  const f16* row = w + (size_t)(nbase + n) * ldw + kbase;
#pragma unroll
  for (int q = 0; q < 8; ++q) {
    const int k0 = (q < 4) ? (hf * 8 + 2 * q) : (16 + hf * 8 + 2 * (q - 4));
    r.u[q] = *(const unsigned*)(row + k0);
  }
  return r.v;
}

__device__ __forceinline__ float sigmoidf_(float v) {
  return 1.0f / (1.0f + __expf(-v));
}

// grid-wide barrier: monotonic counter, target = gen * NWG
__device__ __forceinline__ void grid_barrier(unsigned* cnt, unsigned gen) {
  __syncthreads();
  if (threadIdx.x == 0) {
    __threadfence();
    atomicAdd(cnt, 1u);
    const unsigned target = gen * NWG;
    while (__hip_atomic_load(cnt, __ATOMIC_RELAXED, __HIP_MEMORY_SCOPE_AGENT) < target)
      __builtin_amdgcn_s_sleep(1);
    __threadfence();
  }
  __syncthreads();
}

// out_{tOut} tile for WG g (g<8): [16 batch rows] x [16 out cols], K=512 over a16
__device__ __forceinline__ void out_tile(const f16* __restrict__ a16,
                                         const f16* __restrict__ W216,
                                         const float* __restrict__ b2,
                                         float* __restrict__ out, int tOut, int g, int lane) {
  const int mt2 = g >> 2, ot = g & 3;
  v8f acc = {};
#pragma unroll 4
  for (int kb = 0; kb < 16; ++kb) {
    v16h a = loadA_f16(a16 + (size_t)(16 * mt2) * NHID, NHID, kb * 32, lane);
    v16h b = loadB_f16(W216, NHID, 16 * ot, kb * 32, lane);
    acc = WMMA16(a, b, acc);
  }
  const int n = lane & 15;
  const float bias = b2[16 * ot + n];
#pragma unroll
  for (int r_ = 0; r_ < 8; ++r_) {
    const int b = 16 * mt2 + (lane >> 4) * 8 + r_;
    out[((size_t)b * S_LEN + tOut) * NOUT + 16 * ot + n] = acc[r_] + bias;
  }
}

__global__ void gru_init_kernel(const float* Wz, const float* Wr, const float* Wh,
                                const float* W1, const float* W2, const float* h0,
                                f16* Wz16, f16* Wr16, f16* Wh16, f16* W116, f16* W216,
                                float* hb1, unsigned* cnt) {
  const size_t i = (size_t)blockIdx.x * blockDim.x + threadIdx.x;
  if (i < (size_t)NHID * (NIN + NHID)) {
    Wz16[i] = (f16)Wz[i];
    Wr16[i] = (f16)Wr[i];
    Wh16[i] = (f16)Wh[i];
  }
  if (i < (size_t)NHID * NHID) W116[i] = (f16)W1[i];
  if (i < (size_t)NOUT * NHID) W216[i] = (f16)W2[i];
  if (i < (size_t)BATCH * NHID) hb1[i] = h0[i];
  if (i == 0) *cnt = 0u;
}

__global__ void __launch_bounds__(NTH, 1)
gru_persistent_kernel(const float* __restrict__ x,
                      const float* __restrict__ bz, const float* __restrict__ br,
                      const float* __restrict__ bh, const float* __restrict__ b1,
                      const float* __restrict__ b2,
                      const f16* __restrict__ Wz16, const f16* __restrict__ Wr16,
                      const f16* __restrict__ Wh16, const f16* __restrict__ W116,
                      const f16* __restrict__ W216,
                      float* hbuf /* [2][BATCH*NHID] contiguous ping-pong */,
                      f16* rh16, f16* a16,
                      float* out, unsigned* cnt) {
  const int g    = blockIdx.x;     // 0..15, owns H rows [32g, 32g+32)
  const int tid  = threadIdx.x;
  const int w    = tid >> 5;       // wave 0..7
  const int lane = tid & 31;
  const int n    = lane & 15;      // N index within tile (C/D layout)
  const int hf   = lane >> 4;

  __shared__ float zb[BATCH][32];  // z slice staged for phase B

  unsigned gen = 0;

  // ---------- per-wave constants + weight B-tiles hoisted into registers ----------
  // Phase A: 8 waves -> z tiles (w 0..3), r tiles (w 4..7); (mt,nt) in {0,1}^2
  const int  gate = w >> 2;
  const int  subA = w & 3;
  const int  mtA = subA >> 1, ntA = subA & 1;
  const int  jA  = 32 * g + 16 * ntA + n;
  const float biasA = (gate == 0) ? bz[jA] : br[jA];
  const f16* WgA = (gate == 0) ? Wz16 : Wr16;

  v16h BA[18];                                    // gate weight tile, K=576 (144 VGPRs)
#pragma unroll
  for (int kb = 0; kb < 18; ++kb)
    BA[kb] = loadB_f16(WgA, NIN + NHID, 32 * g + 16 * ntA, kb * 32, lane);

  // Phase B: waves 0-3 -> h_hat tiles (Wh, K=576); waves 4-7 -> a tiles (W1, K=512)
  const int  subB = (w < 4) ? w : (w - 4);
  const int  mtB = subB >> 1, ntB = subB & 1;
  const int  jB  = 32 * g + 16 * ntB + n;
  const float biasB = (w < 4) ? bh[jB] : b1[jB];

  v16h BB[18];                                    // phase-B weight tile (<=144 VGPRs)
  if (w < 4) {
#pragma unroll
    for (int kb = 0; kb < 18; ++kb)
      BB[kb] = loadB_f16(Wh16, NIN + NHID, 32 * g + 16 * ntB, kb * 32, lane);
  } else {
#pragma unroll
    for (int kb = 0; kb < 16; ++kb)
      BB[kb] = loadB_f16(W116, NHID, 32 * g + 16 * ntB, kb * 32, lane);
    BB[16] = BB[0];  // unused, keep defined
    BB[17] = BB[0];
  }

  for (int t = 0; t < S_LEN; ++t) {
    // ping-pong via offsets from ONE global base pointer (keeps address space
    // inference happy -> global_load, not flat_load). h_{-1}=h0 lives at slot 1.
    const float* hprev = hbuf + (size_t)((t + 1) & 1) * HBUF_ELEMS;
    float*       hcur  = hbuf + (size_t)(t & 1) * HBUF_ELEMS;

    // ================= Phase A: z, r gates (+ pipelined out_{t-2}) =================
    {
      v8f acc = {};
      const float* xbase = x + ((size_t)(16 * mtA) * S_LEN + t) * NIN;
#pragma unroll
      for (int kb = 0; kb < 2; ++kb)
        acc = WMMA16(loadA_f32(xbase, XLD, kb * 32, lane), BA[kb], acc);
      const float* hbase = hprev + (size_t)(16 * mtA) * NHID;
#pragma unroll
      for (int kb = 2; kb < 18; ++kb)
        acc = WMMA16(loadA_f32(hbase, NHID, kb * 32 - NIN, lane), BA[kb], acc);

#pragma unroll
      for (int r_ = 0; r_ < 8; ++r_) {
        const int brow = 16 * mtA + hf * 8 + r_;
        const float v = sigmoidf_(acc[r_] + biasA);
        if (gate == 0) {
          zb[brow][16 * ntA + n] = v;
        } else {
          rh16[(size_t)brow * NHID + jA] = (f16)(v * hprev[(size_t)brow * NHID + jA]);
        }
      }
      // prefetch next step's x rows (global_prefetch path); 2 lines per 256B row
      if (t + 1 < S_LEN && tid < 64) {
        const int b = tid & 31, hlf = tid >> 5;
        __builtin_prefetch(x + ((size_t)b * S_LEN + (t + 1)) * NIN + hlf * 32, 0, 3);
      }
    }
    if (w == 0 && g < 8 && t >= 2)
      out_tile(a16, W216, b2, out, t - 2, g, lane);   // out_{t-2} from a_{t-2}

    grid_barrier(cnt, ++gen);  // rh(t) fully visible; a_{t-2} readers done

    // ================= Phase B: h_hat & h_t  ||  a_{t-1} =================
    if (w < 4) {
      // h_hat tile: gate_input = [r*h_prev (512) | x_t (64)]
      v8f acc = {};
      const f16* rbase = rh16 + (size_t)(16 * mtB) * NHID;
#pragma unroll
      for (int kb = 0; kb < 16; ++kb)
        acc = WMMA16(loadA_f16(rbase, NHID, kb * 32, lane), BB[kb], acc);
      const float* xbase = x + ((size_t)(16 * mtB) * S_LEN + t) * NIN;
#pragma unroll
      for (int kb = 16; kb < 18; ++kb)
        acc = WMMA16(loadA_f32(xbase, XLD, kb * 32 - NHID, lane), BB[kb], acc);

#pragma unroll
      for (int r_ = 0; r_ < 8; ++r_) {
        const int brow = 16 * mtB + hf * 8 + r_;
        const float hh = tanhf(acc[r_] + biasB);
        const float zv = zb[brow][16 * ntB + n];
        const float hp = hprev[(size_t)brow * NHID + jB];
        hcur[(size_t)brow * NHID + jB] = (1.0f - zv) * hp + zv * hh;
      }
    } else {
      // a_{t-1} tile: relu(h_{t-1} @ W1^T + b1), h_{t-1} read from hprev (stable)
      v8f acc = {};
      const float* hbase = hprev + (size_t)(16 * mtB) * NHID;
#pragma unroll
      for (int kb = 0; kb < 16; ++kb)
        acc = WMMA16(loadA_f32(hbase, NHID, kb * 32, lane), BB[kb], acc);

#pragma unroll
      for (int r_ = 0; r_ < 8; ++r_) {
        const int brow = 16 * mtB + hf * 8 + r_;
        const float av = fmaxf(acc[r_] + biasB, 0.0f);
        a16[(size_t)brow * NHID + jB] = (f16)av;
      }
    }

    grid_barrier(cnt, ++gen);  // h_t and a_{t-1} fully visible
  }

  // ================= Epilogue: drain pipeline =================
  const float* hlast = hbuf + (size_t)((S_LEN - 1) & 1) * HBUF_ELEMS;

  if (w == 0 && g < 8)
    out_tile(a16, W216, b2, out, S_LEN - 2, g, lane);   // out_{S-2} from a_{S-2}
  grid_barrier(cnt, ++gen);

  // a_{S-1} = relu(h_{S-1} @ W1^T + b1)  (reuses hoisted W1 tiles in BB)
  if (w >= 4) {
    v8f acc = {};
    const float* hbase = hlast + (size_t)(16 * mtB) * NHID;
#pragma unroll
    for (int kb = 0; kb < 16; ++kb)
      acc = WMMA16(loadA_f32(hbase, NHID, kb * 32, lane), BB[kb], acc);
#pragma unroll
    for (int r_ = 0; r_ < 8; ++r_) {
      const int brow = 16 * mtB + hf * 8 + r_;
      a16[(size_t)brow * NHID + jB] = (f16)fmaxf(acc[r_] + biasB, 0.0f);
    }
  }
  // h_last -> d_out tail (each WG writes its own H slice)
  for (int idx = tid; idx < BATCH * 32; idx += NTH) {
    const int b = idx >> 5, jj = idx & 31;
    const int j = 32 * g + jj;
    out[(size_t)BATCH * S_LEN * NOUT + (size_t)b * NHID + j] = hlast[(size_t)b * NHID + j];
  }
  grid_barrier(cnt, ++gen);

  if (w == 0 && g < 8)
    out_tile(a16, W216, b2, out, S_LEN - 1, g, lane);   // out_{S-1}
}

// ---------------- host side ----------------
extern "C" void kernel_launch(void* const* d_in, const int* in_sizes, int n_in,
                              void* d_out, int out_size, void* d_ws, size_t ws_size,
                              hipStream_t stream) {
  const float* x   = (const float*)d_in[0];
  const float* h0  = (const float*)d_in[1];
  const float* Wz  = (const float*)d_in[2];
  const float* Wr  = (const float*)d_in[3];
  const float* Wh  = (const float*)d_in[4];
  const float* bz  = (const float*)d_in[5];
  const float* br  = (const float*)d_in[6];
  const float* bh  = (const float*)d_in[7];
  const float* W1  = (const float*)d_in[8];
  const float* b1  = (const float*)d_in[9];
  const float* W2  = (const float*)d_in[10];
  const float* b2  = (const float*)d_in[11];
  float* out = (float*)d_out;

  char* ws = (char*)d_ws;
  // workspace layout (256B-aligned blocks); HB0/HB1 MUST stay contiguous
  constexpr size_t OFF_CNT = 0;
  constexpr size_t OFF_HB0 = 256;
  constexpr size_t OFF_HB1 = OFF_HB0 + (size_t)HBUF_ELEMS * 4;             // 64KB
  constexpr size_t OFF_RH  = OFF_HB1 + (size_t)HBUF_ELEMS * 4;             // 64KB
  constexpr size_t OFF_A   = OFF_RH  + (size_t)BATCH * NHID * 2;           // 32KB
  constexpr size_t OFF_WZ  = OFF_A   + (size_t)BATCH * NHID * 2;           // 32KB
  constexpr size_t OFF_WR  = OFF_WZ  + (size_t)NHID * (NIN + NHID) * 2;
  constexpr size_t OFF_WH  = OFF_WR  + (size_t)NHID * (NIN + NHID) * 2;
  constexpr size_t OFF_W1  = OFF_WH  + (size_t)NHID * (NIN + NHID) * 2;
  constexpr size_t OFF_W2  = OFF_W1  + (size_t)NHID * NHID * 2;

  unsigned* cnt  = (unsigned*)(ws + OFF_CNT);
  float*    hb0  = (float*)(ws + OFF_HB0);
  float*    hb1  = (float*)(ws + OFF_HB1);
  f16*      rh16 = (f16*)(ws + OFF_RH);
  f16*      a16  = (f16*)(ws + OFF_A);
  f16*      Wz16 = (f16*)(ws + OFF_WZ);
  f16*      Wr16 = (f16*)(ws + OFF_WR);
  f16*      Wh16 = (f16*)(ws + OFF_WH);
  f16*      W116 = (f16*)(ws + OFF_W1);
  f16*      W216 = (f16*)(ws + OFF_W2);

  const int init_blocks = (NHID * (NIN + NHID) + NTH - 1) / NTH;  // covers largest array
  gru_init_kernel<<<init_blocks, NTH, 0, stream>>>(Wz, Wr, Wh, W1, W2, h0,
                                                   Wz16, Wr16, Wh16, W116, W216,
                                                   hb1, cnt);
  gru_persistent_kernel<<<NWG, NTH, 0, stream>>>(x, bz, br, bh, b1, b2,
                                                 Wz16, Wr16, Wh16, W116, W216,
                                                 hb0, rh16, a16, out, cnt);
}
